// AdditiveAttention_32427003085517
// MI455X (gfx1250) — matrix operand order repaired and checked by reading, not compiled, not run on hardware
//
#include <hip/hip_runtime.h>
#include <hip/hip_bf16.h>

// ---------------------------------------------------------------------------
// Additive attention, MI455X (gfx1250, wave32, WMMA).
//   B=16, NK=NQ=256, DK=DQ=256, H=128, DV=256
// Kernel 1: kproj = key@Wk, qproj = query@Wq   (WMMA f16, f32 accum)
// Kernel 2: fused tanh-scores -> masked softmax -> attn@value (WMMA)
// ---------------------------------------------------------------------------

typedef __attribute__((ext_vector_type(16))) _Float16 v16h;
typedef __attribute__((ext_vector_type(8)))  float    v8f;

#define AB_B    16
#define AB_NK   256
#define AB_NQ   256
#define AB_DK   256
#define AB_H    128
#define AB_DV   256
#define AB_MASK (-1000000.0f)

// Single-instruction v_tanh_f32 (native TRANS op on gfx1250), with an
// inline-asm fallback that guarantees the 1-instruction path even if the
// clang builtin is named differently in this toolchain.
__device__ __forceinline__ float fast_tanh(float x) {
#if __has_builtin(__builtin_amdgcn_tanhf)
    return __builtin_amdgcn_tanhf(x);
#elif __has_builtin(__builtin_amdgcn_tanh_f32)
    return __builtin_amdgcn_tanh_f32(x);
#else
    float y;
    asm("v_tanh_f32 %0, %1" : "=v"(y) : "v"(x));
    return y;
#endif
}

// ---------------------------------------------------------------------------
// Kernel 1: projections.  4096 wave-tiles total:
//   tiles [0,2048)    -> kproj = key  (4096x256) @ Wk (256x128)
//   tiles [2048,4096) -> qproj = query(4096x256) @ Wq (256x128)
// One 16x16 output tile per wave, K = 256 in 8 steps of 32.
// ---------------------------------------------------------------------------
__global__ __launch_bounds__(256) void ab_proj_kernel(
    const float* __restrict__ key, const float* __restrict__ query,
    const float* __restrict__ Wk,  const float* __restrict__ Wq,
    float* __restrict__ kproj,     float* __restrict__ qproj)
{
    const int lane = threadIdx.x & 31;
    const int wave = threadIdx.x >> 5;
    int tile = blockIdx.x * 8 + wave;              // 0..4095

    const float* A; const float* W; float* D;
    if (tile < 2048) { A = key;   W = Wk; D = kproj; }
    else             { A = query; W = Wq; D = qproj; tile -= 2048; }

    const int mt = tile >> 3;                      // 0..255 row tile
    const int nt = tile & 7;                       // 0..7   col tile
    const int m0 = mt * 16, n0 = nt * 16;

    const int half = lane >> 4;                    // 0 or 1
    const int Mr   = lane & 15;                    // A-frag row
    const int kbA  = half * 8;                     // A-frag K sub-base
    const int N    = n0 + (lane & 15);             // B/C/D column
    const int kbB  = half * 16;                    // B-frag K sub-base

    const float* arow = A + (size_t)(m0 + Mr) * AB_DK;

    v8f c = {};
    #pragma unroll 2
    for (int kk = 0; kk < AB_DK; kk += 32) {
        // A 16x32 f16 layout: lane covers K = kb+[0..7] U kb+[16..23],
        // both contiguous 8-float runs -> 4 x b128 loads.
        const float4 f0 = *reinterpret_cast<const float4*>(arow + kk + kbA);
        const float4 f1 = *reinterpret_cast<const float4*>(arow + kk + kbA + 4);
        const float4 f2 = *reinterpret_cast<const float4*>(arow + kk + kbA + 16);
        const float4 f3 = *reinterpret_cast<const float4*>(arow + kk + kbA + 20);
        v16h a, bw;
        a[0]=(_Float16)f0.x; a[1]=(_Float16)f0.y; a[2]=(_Float16)f0.z; a[3]=(_Float16)f0.w;
        a[4]=(_Float16)f1.x; a[5]=(_Float16)f1.y; a[6]=(_Float16)f1.z; a[7]=(_Float16)f1.w;
        a[8]=(_Float16)f2.x; a[9]=(_Float16)f2.y; a[10]=(_Float16)f2.z; a[11]=(_Float16)f2.w;
        a[12]=(_Float16)f3.x; a[13]=(_Float16)f3.y; a[14]=(_Float16)f3.z; a[15]=(_Float16)f3.w;
        #pragma unroll
        for (int i = 0; i < 16; ++i) {
            // B 32x16 f16 layout: lane half h covers K = 8h+[0..7] U 8h+[16..23]
            const int kB = kk + kbA + (i < 8 ? i : i + 8);
            bw[i] = (_Float16)W[(size_t)kB * AB_H + N];
        }
        c = __builtin_amdgcn_wmma_f32_16x16x32_f16(
                false, a, false, bw, (short)0, c, false, false);
    }

    #pragma unroll
    for (int r = 0; r < 8; ++r) {
        const int M = r + half * 8;                // C/D layout
        D[(size_t)(m0 + M) * AB_H + N] = c[r];
    }
}

// ---------------------------------------------------------------------------
// Kernel 2: one block per (batch, 16-key-row tile).  256 threads = 8 waves.
//  Phase 1: thread t owns query column q=t; 16 tanh-scores vs the staged
//           16x128 kproj tile.  Masked columns (q >= valid_len) skip all work.
//  Phase 2: per-row softmax over q (2 rows per wave, shfl reductions).
//  Phase 3: out[16,256] = attn[16,256] @ value[b] via WMMA, 2 DV tiles/wave.
// ---------------------------------------------------------------------------
__global__ __launch_bounds__(256) void ab_attn_kernel(
    const float* __restrict__ value, const int* __restrict__ valid_lens,
    const float* __restrict__ wv,    const float* __restrict__ kproj,
    const float* __restrict__ qproj, float* __restrict__ out)
{
    __shared__ float s_kp[16][AB_H];       // kproj tile (broadcast reads)
    __shared__ float s_sc[16][AB_NQ + 1];  // scores/attn, padded stride 257
    __shared__ float s_wv[AB_H];

    const int t    = threadIdx.x;
    const int lane = t & 31;
    const int wave = t >> 5;
    const int bat  = blockIdx.x >> 4;      // batch
    const int kt   = blockIdx.x & 15;      // key-row tile
    const int k0   = kt * 16;
    const int vl   = valid_lens[bat];

    const float* vb = value + (size_t)bat * AB_NQ * AB_DV;

    // Warm L2/WGP$ for the value[b] slab this block re-reads in phase 3:
    // 256 threads x 8 cachelines (128B) = 256 KB.
    {
        const char* vbase = (const char*)(vb) + (size_t)t * 1024;
        #pragma unroll
        for (int j = 0; j < 8; ++j)
            __builtin_prefetch(vbase + j * 128, 0, 3);
    }

    // ---- stage kproj tile + wv ----
    {
        const int r  = t >> 4;             // 0..15
        const int h0 = (t & 15) * 8;
        const float* src = kproj + ((size_t)(bat * AB_NK + k0 + r)) * AB_H + h0;
        const float4 p0 = *reinterpret_cast<const float4*>(src);
        const float4 p1 = *reinterpret_cast<const float4*>(src + 4);
        *reinterpret_cast<float4*>(&s_kp[r][h0])     = p0;
        *reinterpret_cast<float4*>(&s_kp[r][h0 + 4]) = p1;
        if (t < AB_H) s_wv[t] = wv[t];
    }
    __syncthreads();

    // ---- phase 1: scores (tanh) ----
    {
        const int q = t;
        if (q < vl) {
            float acc[16];
            #pragma unroll
            for (int k = 0; k < 16; ++k) acc[k] = 0.f;
            const float* qrow = qproj + ((size_t)(bat * AB_NQ + q)) * AB_H;
            for (int h0 = 0; h0 < AB_H; h0 += 32) {
                float qv[32], wvv[32];
                #pragma unroll
                for (int j = 0; j < 8; ++j) {
                    const float4 f = *reinterpret_cast<const float4*>(qrow + h0 + 4 * j);
                    qv[4*j]   = f.x; qv[4*j+1] = f.y;
                    qv[4*j+2] = f.z; qv[4*j+3] = f.w;
                }
                #pragma unroll
                for (int j = 0; j < 32; ++j) wvv[j] = s_wv[h0 + j];
                for (int k = 0; k < 16; ++k) {
                    float a = 0.f;
                    #pragma unroll
                    for (int j = 0; j < 32; ++j)
                        a += wvv[j] * fast_tanh(s_kp[k][h0 + j] + qv[j]);
                    acc[k] += a;
                }
            }
            #pragma unroll
            for (int k = 0; k < 16; ++k) s_sc[k][q] = acc[k];
        } else {
            #pragma unroll
            for (int k = 0; k < 16; ++k) s_sc[k][q] = AB_MASK;
        }
    }
    __syncthreads();

    // ---- phase 2: row softmax over q (wave handles rows 2w, 2w+1) ----
    #pragma unroll
    for (int sub = 0; sub < 2; ++sub) {
        const int r = wave * 2 + sub;
        float v[8];
        float m = -3.0e38f;
        #pragma unroll
        for (int i = 0; i < 8; ++i) {
            v[i] = s_sc[r][lane + 32 * i];
            m = fmaxf(m, v[i]);
        }
        #pragma unroll
        for (int off = 16; off > 0; off >>= 1)
            m = fmaxf(m, __shfl_xor(m, off, 32));
        float sum = 0.f;
        #pragma unroll
        for (int i = 0; i < 8; ++i) { v[i] = __expf(v[i] - m); sum += v[i]; }
        #pragma unroll
        for (int off = 16; off > 0; off >>= 1)
            sum += __shfl_xor(sum, off, 32);
        const float inv = __builtin_amdgcn_rcpf(sum);
        #pragma unroll
        for (int i = 0; i < 8; ++i) s_sc[r][lane + 32 * i] = v[i] * inv;
    }
    __syncthreads();

    // ---- phase 3: out = attn @ value[b] (WMMA), 2 column tiles per wave ----
    {
        const int half = lane >> 4;
        const int Mr   = lane & 15;
        const int kbA  = half * 8;
        const int kbB  = half * 16;
        const int n0   = wave * 32;

        v8f c0 = {}, c1 = {};
        #pragma unroll 2
        for (int kk = 0; kk < AB_NQ; kk += 32) {
            v16h a, b0, b1;
            #pragma unroll
            for (int i = 0; i < 16; ++i)
                a[i] = (_Float16)s_sc[Mr][kk + kbA + (i < 8 ? i : i + 8)];
            #pragma unroll
            for (int i = 0; i < 16; ++i) {
                const float* row = vb + (size_t)(kk + kbA + (i < 8 ? i : i + 8)) * AB_DV;
                b0[i] = (_Float16)row[n0 + (lane & 15)];
                b1[i] = (_Float16)row[n0 + 16 + (lane & 15)];
            }
            c0 = __builtin_amdgcn_wmma_f32_16x16x32_f16(
                     false, a, false, b0, (short)0, c0, false, false);
            c1 = __builtin_amdgcn_wmma_f32_16x16x32_f16(
                     false, a, false, b1, (short)0, c1, false, false);
        }
        #pragma unroll
        for (int r = 0; r < 8; ++r) {
            const int M = r + half * 8;
            float* orow = out + ((size_t)(bat * AB_NK + k0 + M)) * AB_DV;
            orow[n0 + (lane & 15)]      = c0[r];
            orow[n0 + 16 + (lane & 15)] = c1[r];
        }
    }
}

// ---------------------------------------------------------------------------
extern "C" void kernel_launch(void* const* d_in, const int* in_sizes, int n_in,
                              void* d_out, int out_size, void* d_ws, size_t ws_size,
                              hipStream_t stream) {
    const float* key        = (const float*)d_in[0];
    const float* query      = (const float*)d_in[1];
    const float* value      = (const float*)d_in[2];
    const int*   valid_lens = (const int*)d_in[3];
    const float* Wk         = (const float*)d_in[4];
    const float* Wq         = (const float*)d_in[5];
    const float* wv         = (const float*)d_in[6];
    float* out = (float*)d_out;

    float* kproj = (float*)d_ws;                        // 16*256*128 f32 = 2 MB
    float* qproj = kproj + (size_t)AB_B * AB_NK * AB_H; // next 2 MB

    // 4096 wave-tiles / 8 waves per block = 512 blocks
    ab_proj_kernel<<<512, 256, 0, stream>>>(key, query, Wk, Wq, kproj, qproj);
    // one block per (batch, 16-key-row tile) = 16*16 = 256 blocks
    ab_attn_kernel<<<256, 256, 0, stream>>>(value, valid_lens, wv, kproj, qproj, out);
}
